// FHNet_62397284876289
// MI455X (gfx1250) — matrix operand order repaired and checked by reading, not compiled, not run on hardware
//
#include <hip/hip_runtime.h>

// ---------------------------------------------------------------------------
// FHNet on MI455X (gfx1250, wave32, WMMA bf16 16x16x32)
//
//   kernel 1: gemm1_wmma : z1[b,t,h] = batch[b,t,:] . W1[h,:]  (bf16 in ws)
//             software-pipelined staging with explicit sched_group_barrier
//             ordering: VMEM prefetch -> 16 DS reads -> 7 WMMAs -> cvt -> DS writes
//   kernel 2: fhn_fused  : 3-stage wave-specialized pipeline
//             waves 0-3: scan1(c) | waves 4-6: GEMM2(c-1) | wave 7: scan2(c-2)
// ---------------------------------------------------------------------------

typedef __bf16 bf16;
typedef __bf16 v2bf  __attribute__((ext_vector_type(2)));
typedef __bf16 v8bf  __attribute__((ext_vector_type(8)));
typedef __bf16 v16bf __attribute__((ext_vector_type(16)));
typedef float  v8f   __attribute__((ext_vector_type(8)));
typedef float  v4f   __attribute__((ext_vector_type(4)));

#define FHN_DT 0.01f

#define BATCH  100
#define TSTEPS 2000
#define DIN    784
#define HID    100
#define HP     112                // HID padded to 7*16 for WMMA n-tiles
#define NOUT   10
#define M_TOT  (BATCH * TSTEPS)   // 200000 rows

// ---- GEMM1 tiling ----
#define BM   128                  // rows (t) per block -> 8 waves x 16
#define BK   32                   // k slab (bf16 WMMA K)
#define NT   7                    // n tiles of 16 (HP = 112)
#define LSTR 40                   // LDS row stride in bf16 (80B, bank-conflict pad)
#define NBP  7                    // B bf16-pairs per thread per slab: 112*16/256

// sched_group_barrier masks
#define SGB_VALU   0x002
#define SGB_WMMA   0x008
#define SGB_VMREAD 0x020
#define SGB_DSREAD 0x100
#define SGB_DSWRITE 0x200

__global__ __launch_bounds__(256)
__attribute__((amdgpu_waves_per_eu(1)))
void gemm1_wmma(const float* __restrict__ batch,
                const float* __restrict__ W1,
                bf16* __restrict__ z1)
{
    __shared__ bf16 As[2][BM][LSTR];   // A slab: 128 x 32 bf16
    __shared__ bf16 Bs[2][HP][LSTR];   // B slab (transposed): Bs[h][kk] = W1[h][k0+kk]

    const int  tid  = threadIdx.x;
    const int  w    = tid >> 5;        // wave 0..7, owns m-tile rows [w*16, w*16+16)
    const int  l    = tid & 31;
    const int  m    = l & 15;          // A row-in-tile == B column lane
    const int  half = l >> 4;          // selects K groups per ISA fragment layout
    const long m0   = (long)blockIdx.x * BM;

    v8f acc[NT] = {};

    float4 av[4];          // prefetched A slab piece (16 f32 / thread)
    float2 bv[NBP];        // prefetched B slab piece (14 f32 / thread, as pairs)
    bool   aok[4];
    bool   bok[NBP];

    // Branch-free prefetch: clamp addresses in-bounds, remember validity,
    // zero-select after the (always legal) load.
    auto load_global = [&](int ks) {
        const int k0 = ks * BK;
#pragma unroll
        for (int i = 0; i < 4; ++i) {
            int  row = i * 32 + (tid >> 3);
            int  col = (tid & 7) << 2;
            long g   = m0 + row;
            aok[i]   = (g < (long)M_TOT) && (k0 + col) < DIN;
            long gc  = g < (long)M_TOT ? g : (long)(M_TOT - 1);
            int  kc  = (k0 + col) < DIN ? (k0 + col) : 0;
            av[i] = *(const float4*)(batch + gc * (long)DIN + kc);
        }
#pragma unroll
        for (int j = 0; j < NBP; ++j) {
            int idx = tid + j * 256;       // pair index, 16 pairs per h row
            int h   = idx >> 4;
            int k   = k0 + ((idx & 15) << 1);
            bok[j]  = (h < HID) && (k < DIN);
            int hc  = h < HID ? h : (HID - 1);
            int kc  = k < DIN ? k : 0;
            bv[j] = *(const float2*)(W1 + hc * DIN + kc);
        }
    };

    auto store_lds = [&](int buf) {
#pragma unroll
        for (int i = 0; i < 4; ++i) {
            int row = i * 32 + (tid >> 3);
            int col = (tid & 7) << 2;
            float4 v = av[i];
            if (!aok[i]) v = make_float4(0.f, 0.f, 0.f, 0.f);   // cndmask
            As[buf][row][col + 0] = (bf16)v.x;
            As[buf][row][col + 1] = (bf16)v.y;
            As[buf][row][col + 2] = (bf16)v.z;
            As[buf][row][col + 3] = (bf16)v.w;
        }
#pragma unroll
        for (int j = 0; j < NBP; ++j) {
            int idx = tid + j * 256;
            float2 v = bv[j];
            if (!bok[j]) v = make_float2(0.f, 0.f);              // cndmask
            v2bf p = { (bf16)v.x, (bf16)v.y };
            *(v2bf*)&Bs[buf][idx >> 4][(idx & 15) << 1] = p;     // packed b32 store
        }
    };

    auto compute = [&](int buf) {
        // A fragment (16x32): lanes 0-15 hold K{0..7,16..23}, lanes 16-31 K{8..15,24..31}
        const bf16* ap = &As[buf][w * 16 + m][half * 8];
        v8bf alo = *(const v8bf*)ap;
        v8bf ahi = *(const v8bf*)(ap + 16);
        v16bf afrag = __builtin_shufflevector(alo, ahi,
            0,1,2,3,4,5,6,7,8,9,10,11,12,13,14,15);
        v16bf bfrag[NT];
#pragma unroll
        for (int nt = 0; nt < NT; ++nt) {
            const bf16* bp = &Bs[buf][nt * 16 + m][half * 8];
            v8bf blo = *(const v8bf*)bp;
            v8bf bhi = *(const v8bf*)(bp + 16);
            bfrag[nt] = __builtin_shufflevector(blo, bhi,
                0,1,2,3,4,5,6,7,8,9,10,11,12,13,14,15);
        }
#pragma unroll
        for (int nt = 0; nt < NT; ++nt)
            acc[nt] = __builtin_amdgcn_wmma_f32_16x16x32_bf16(
                false, afrag, false, bfrag[nt], (short)0, acc[nt], false, false);
    };

    const int NK = (DIN + BK - 1) / BK;   // 25 slabs (last half-slab zero padded)
    load_global(0);
    store_lds(0);
    __syncthreads();
    for (int ks = 0; ks < NK; ++ks) {
        if (ks + 1 < NK) load_global(ks + 1);     // issue HBM loads (no wait yet)
        compute(ks & 1);                          // LDS + WMMA overlaps load latency
        if (ks + 1 < NK) store_lds((ks + 1) & 1); // wait, convert, stage

        // Pin the per-slab pipeline order (best-effort, deps respected):
        //   1) issue all 11 HBM prefetch loads
        //   2) all 16 fragment DS reads (distinct regs -> partial dscnt waits)
        //   3) 7 WMMAs back-to-back
        //   4) conversion/select VALU (pulls s_wait_loadcnt after the WMMAs)
        //   5) staging DS writes
        __builtin_amdgcn_sched_group_barrier(SGB_VMREAD, 11, 0);
        __builtin_amdgcn_sched_group_barrier(SGB_DSREAD, 16, 0);
        __builtin_amdgcn_sched_group_barrier(SGB_WMMA,    7, 0);
        __builtin_amdgcn_sched_group_barrier(SGB_VALU,   64, 0);
        __builtin_amdgcn_sched_group_barrier(SGB_DSWRITE, 16, 0);
        __syncthreads();
    }

    // C layout: lane l, VGPR r -> (M = half*8 + r, N = m)
    const long rowbase = m0 + w * 16 + half * 8;
#pragma unroll
    for (int nt = 0; nt < NT; ++nt) {
        const int col = nt * 16 + m;
#pragma unroll
        for (int r = 0; r < 8; ++r) {
            long row = rowbase + r;
            if (row < (long)M_TOT)
                z1[row * HP + col] = (bf16)acc[nt][r];
        }
    }
}

// ---------------------------------------------------------------------------
// Fused FHN: 3-stage pipeline over chunks of TC timesteps.
//   stage 1 (waves 0-3, lanes h<100): scan1(c)   -> v1buf[c&1]
//   stage 2 (waves 4-6, 96 threads) : GEMM2(c-1) -> z2buf[(c-1)&1]
//   stage 3 (wave 7, lanes o<10)    : scan2(c-2) -> out
// One workgroup barrier per chunk.
// ---------------------------------------------------------------------------
#define TC 50
#define NC (TSTEPS / TC)   // 40 chunks

__global__ __launch_bounds__(256)
void fhn_fused(const bf16* __restrict__ z1,
               const float* __restrict__ W2,
               float* __restrict__ out)
{
    __shared__ __align__(16) float w2s[NOUT][104];
    __shared__ __align__(16) float v1buf[2][TC][104];
    __shared__ __align__(16) float z2buf[2][TC][12];

    const int tid = threadIdx.x;
    const int b   = blockIdx.x;

    for (int e = tid; e < NOUT * 104; e += 256) {
        int o = e / 104, h = e % 104;
        w2s[o][h] = (h < HID) ? W2[o * HID + h] : 0.f;
    }

    float V1 = 0.f, S1 = 0.f;   // scan1 carry (tid < HID owns h = tid)
    float V2 = 0.f, S2 = 0.f;   // scan2 carry (tid-224 < NOUT owns o)
    const long zbase = (long)b * TSTEPS * HP;
    const long obase = (long)b * TSTEPS * NOUT;
    __syncthreads();

    for (int c = 0; c < NC + 2; ++c) {
        if (tid < 128) {
            // ---- stage 1: scan1 chunk c ----
            if (tid < HID && c < NC) {
                const int   h  = tid;
                const int   t0 = c * TC;
                const bf16* zp = z1 + zbase + (long)t0 * HP + h;
                float* vb = &v1buf[c & 1][0][h];
#pragma unroll 5
                for (int tt = 0; tt < TC; ++tt) {
                    vb[tt * 104] = V1;                   // V1 holds V[t0+tt]
                    float z  = (float)zp[(long)tt * HP]; // z1[t] drives step t->t+1
                    float V3 = V1 * V1 * V1;
                    float Vn = fmaf(1.f + FHN_DT, V1,
                               fmaf(-(FHN_DT / 3.f), V3,
                               fmaf(FHN_DT, z, -FHN_DT * S1)));
                    float Sn = fmaf(FHN_DT * 0.08f, V1 + 0.7f - 0.8f * S1, S1);
                    V1 = Vn; S1 = Sn;
                }
            }
        } else if (tid < 224) {
            // ---- stage 2: GEMM2 chunk c-1 ----
            if (c >= 1 && c <= NC) {
                const int c1 = c - 1;
                for (int p = tid - 128; p < TC * NOUT; p += 96) {
                    int tt = p / NOUT, o = p % NOUT;
                    const v4f* vp = (const v4f*)&v1buf[c1 & 1][tt][0];
                    const v4f* wp = (const v4f*)&w2s[o][0];
                    v4f a4 = {0.f, 0.f, 0.f, 0.f};
#pragma unroll
                    for (int i = 0; i < HID / 4; ++i)
                        a4 += vp[i] * wp[i];             // v_pk_fma_f32 candidates
                    z2buf[c1 & 1][tt][o] = 0.5f * (a4.x + a4.y + a4.z + a4.w);
                }
            }
        } else {
            // ---- stage 3: scan2 chunk c-2 ----
            if (tid < 224 + NOUT && c >= 2) {
                const int c2 = c - 2;
                const int o  = tid - 224;
                const int t0 = c2 * TC;
                const float* zb = &z2buf[c2 & 1][0][o];
                float* op = out + obase + (long)t0 * NOUT + o;
#pragma unroll 5
                for (int tt = 0; tt < TC; ++tt) {
                    op[tt * NOUT] = V2;                  // V2 holds V[t0+tt]
                    float z  = zb[tt * 12];
                    float V3 = V2 * V2 * V2;
                    float Vn = fmaf(1.f + FHN_DT, V2,
                               fmaf(-(FHN_DT / 3.f), V3,
                               fmaf(FHN_DT, z, -FHN_DT * S2)));
                    float Sn = fmaf(FHN_DT * 0.08f, V2 + 0.7f - 0.8f * S2, S2);
                    V2 = Vn; S2 = Sn;
                }
            }
        }
        __syncthreads();
    }
}

// ---------------------------------------------------------------------------
extern "C" void kernel_launch(void* const* d_in, const int* in_sizes, int n_in,
                              void* d_out, int out_size, void* d_ws, size_t ws_size,
                              hipStream_t stream)
{
    const float* batch = (const float*)d_in[0];   // (100, 2000, 784) f32
    const float* W1    = (const float*)d_in[1];   // (100, 784) f32
    const float* W2    = (const float*)d_in[2];   // (10, 100) f32
    float*       out   = (float*)d_out;           // (100, 2000, 10) f32
    bf16*        z1    = (bf16*)d_ws;             // 200000 x 112 bf16 = 44.8 MB

    const int grid1 = (M_TOT + BM - 1) / BM;      // 1563 blocks
    gemm1_wmma<<<grid1, 256, 0, stream>>>(batch, W1, z1);
    fhn_fused<<<BATCH, 256, 0, stream>>>(z1, W2, out);
}
